// BiGRU_29901562314941
// MI455X (gfx1250) — matrix operand order repaired and checked by reading, not compiled, not run on hardware
//
#include <hip/hip_runtime.h>
#include <hip/hip_bf16.h>
#include <cmath>

// Problem constants (from reference)
#define NB_   512
#define MS_   8
#define T_    64
#define IN_   360
#define H_    230
#define OUT_  53
#define DOCS_ 32
#define ENT_  8
#define B_    (NB_ * MS_)   // 4096 sequences
#define H2_   (2 * H_)      // 460
#define H3_   (3 * H_)      // 690

// hs concat row: [hf (230) pad2 | hr (230) pad2] -> stride 464 bf16 (928 B, 16B-aligned segs)
#define HSEG_ 232
#define HRS_  464

typedef __attribute__((ext_vector_type(16))) __bf16 v16bf;
typedef __attribute__((ext_vector_type(8)))  __bf16 v8bf;
typedef __attribute__((ext_vector_type(8)))  float  v8f;

// LDS leading dims: stride 40 bf16 (80 B) per row/col.
// => per-lane bank stride 20 words (distinct banks for 16 lanes),
//    every 8-element group is 16-byte aligned (b128 LDS ops).
#define ALD 40
#define BLD 40

// M-tile = 128 rows (4 waves x 32 rows, 2 row-subtiles of 16 each)
#define MT 128

__device__ __forceinline__ v8f vzero8() {
  v8f z;
#pragma unroll
  for (int i = 0; i < 8; ++i) z[i] = 0.f;
  return z;
}

__device__ __forceinline__ void store8(__bf16* dst, const float* v) {
  v8bf t;
#pragma unroll
  for (int i = 0; i < 8; ++i) t[i] = (__bf16)v[i];
  *(v8bf*)dst = t;   // single ds_store_b128
}

// A fragment 16x32 bf16 from As[MT][ALD] (row-linear K):
// lane = half*16+m holds row m; element e -> K = (e>>3)*16 + half*8 + (e&7)
__device__ __forceinline__ v16bf frag_a_lds(const __bf16* As, int row0) {
  const int lane = threadIdx.x & 31;
  const int half = lane >> 4, m = lane & 15;
  const v8bf* p = (const v8bf*)(As + (row0 + m) * ALD + half * 8);
  v8bf lo = p[0];   // K = half*8 + 0..7
  v8bf hi = p[2];   // K = 16 + half*8 + 0..7
  return __builtin_shufflevector(lo, hi, 0,1,2,3,4,5,6,7,8,9,10,11,12,13,14,15);
}

// B fragment 32x16 bf16 from Bs[col*BLD + k]
// lane = half*16+n holds col n0+n; element e -> K = (e>>3)*16 + half*8 + (e&7)
__device__ __forceinline__ v16bf frag_b_lds(const __bf16* Bs, int n0) {
  const int lane = threadIdx.x & 31;
  const int half = lane >> 4, n = lane & 15;
  const v8bf* p = (const v8bf*)(Bs + (n0 + n) * BLD + half * 8);
  v8bf lo = p[0];
  v8bf hi = p[2];
  return __builtin_shufflevector(lo, hi, 0,1,2,3,4,5,6,7,8,9,10,11,12,13,14,15);
}

__device__ __forceinline__ v8f wmma_bf16(v16bf a, v16bf b, v8f c) {
  return __builtin_amdgcn_wmma_f32_16x16x32_bf16(false, a, false, b, (short)0, c, false, false);
}

// ---------------------------------------------------------------------------
// Per-timestep GRU GEMM (both directions via blockIdx.y).
// A row b = [ h[b] (230) | bag[b, tt, :] (360) ], K=590 padded to 608.
// N-tiles 0..10 : combined = h@Whh^T + x@Wih^T + bhh + bih      (N=690)
// N-tiles 11..14: xn       = x@Wih_n^T + bih_n                  (N=230)
// ---------------------------------------------------------------------------
__global__ __launch_bounds__(128)
void gru_step_gemm(const float* __restrict__ bag,
                   const float* __restrict__ Whh_f, const float* __restrict__ Wih_f,
                   const float* __restrict__ bhh_f, const float* __restrict__ bih_f,
                   const float* __restrict__ Whh_r, const float* __restrict__ Wih_r,
                   const float* __restrict__ bhh_r, const float* __restrict__ bih_r,
                   const float* __restrict__ h_f, const float* __restrict__ h_r,
                   float* __restrict__ hg_f, float* __restrict__ hg_r,
                   float* __restrict__ xn_f, float* __restrict__ xn_r,
                   int t)
{
  __shared__ __bf16 As[MT * ALD];
  __shared__ __bf16 Bs[64 * BLD];

  const int dir  = blockIdx.y;
  const int row0 = blockIdx.x * MT;
  const int tid  = threadIdx.x;
  const int wave = tid >> 5;
  const int lane = tid & 31, half = lane >> 4, ncol = lane & 15;

  const float* Whh = dir ? Whh_r : Whh_f;
  const float* Wih = dir ? Wih_r : Wih_f;
  const float* bhh = dir ? bhh_r : bhh_f;
  const float* bih = dir ? bih_r : bih_f;
  const float* h   = dir ? h_r   : h_f;
  float* hg        = dir ? hg_r  : hg_f;
  float* xn        = dir ? xn_r  : xn_f;
  const int tt = dir ? (T_ - 1 - t) : t;
  const float* xrow = bag + (size_t)tt * IN_;   // row b -> + b*T*IN

  const int KP = 608;  // 590 padded

  for (int nt = 0; nt < 15; ++nt) {
    const bool isXN = (nt >= 11);
    const int n0   = isXN ? (nt - 11) * 64 : nt * 64;
    const int Nlim = isXN ? H_ : H3_;
    const bool fullN = (n0 + 64 <= Nlim);
    v8f acc[2][4];
#pragma unroll
    for (int s = 0; s < 2; ++s)
#pragma unroll
      for (int j = 0; j < 4; ++j) acc[s][j] = vzero8();

    const int k0start = isXN ? 224 : 0;  // xn uses only x segment (k>=230)
    for (int k0 = k0start; k0 < KP; k0 += 32) {
      // --- stage A chunk: MT rows x 4 groups of 8, scalar group branches ---
#pragma unroll
      for (int it = 0; it < 4; ++it) {
        const int r = tid;                 // 0..127 rows
        const int gk0 = k0 + it * 8;       // scalar
        const int row = row0 + r;
        float v[8];
        if (gk0 + 8 <= H_) {               // pure h segment
#pragma unroll
          for (int i = 0; i < 8; ++i) v[i] = h[(size_t)row * H_ + gk0 + i];
        } else if (gk0 >= H_ && gk0 + 8 <= H_ + IN_) {   // pure x segment
#pragma unroll
          for (int i = 0; i < 8; ++i) v[i] = xrow[(size_t)row * (T_ * IN_) + (gk0 - H_) + i];
        } else if (gk0 >= H_ + IN_) {      // pure zero pad
#pragma unroll
          for (int i = 0; i < 8; ++i) v[i] = 0.f;
        } else {                            // rare boundary-mixed group
#pragma unroll
          for (int i = 0; i < 8; ++i) {
            int gk = gk0 + i;
            float x = 0.f;
            if (gk < H_)            x = h[(size_t)row * H_ + gk];
            else if (gk < H_ + IN_) x = xrow[(size_t)row * (T_ * IN_) + (gk - H_)];
            v[i] = x;
          }
        }
        store8(&As[r * ALD + it * 8], v);
      }
      // --- stage B chunk (weights are N x K, i.e. x @ W^T) ---
#pragma unroll
      for (int it = 0; it < 2; ++it) {
        const int n = tid & 63;
        const int kg = __builtin_amdgcn_readfirstlane((tid >> 6) + it * 2);
        const int gk0 = k0 + kg * 8;       // scalar
        const int gn = n0 + n;
        float v[8];
        if (!isXN) {
          if (fullN && gk0 + 8 <= H_) {
#pragma unroll
            for (int i = 0; i < 8; ++i) v[i] = Whh[(size_t)gn * H_ + gk0 + i];
          } else if (fullN && gk0 >= H_ && gk0 + 8 <= H_ + IN_) {
#pragma unroll
            for (int i = 0; i < 8; ++i) v[i] = Wih[(size_t)gn * IN_ + (gk0 - H_) + i];
          } else {
#pragma unroll
            for (int i = 0; i < 8; ++i) {
              int gk = gk0 + i;
              float x = 0.f;
              if (gn < H3_) {
                if (gk < H_)            x = Whh[(size_t)gn * H_ + gk];
                else if (gk < H_ + IN_) x = Wih[(size_t)gn * IN_ + (gk - H_)];
              }
              v[i] = x;
            }
          }
        } else {
          if (fullN && gk0 >= H_ && gk0 + 8 <= H_ + IN_) {
#pragma unroll
            for (int i = 0; i < 8; ++i) v[i] = Wih[(size_t)(H2_ + gn) * IN_ + (gk0 - H_) + i];
          } else {
#pragma unroll
            for (int i = 0; i < 8; ++i) {
              int gk = gk0 + i;
              float x = 0.f;
              if (gn < H_ && gk >= H_ && gk < H_ + IN_)
                x = Wih[(size_t)(H2_ + gn) * IN_ + (gk - H_)];
              v[i] = x;
            }
          }
        }
        store8(&Bs[n * BLD + kg * 8], v);
      }
      __syncthreads();
      v16bf a0 = frag_a_lds(As, wave * 32);
      v16bf a1 = frag_a_lds(As, wave * 32 + 16);
#pragma unroll
      for (int j = 0; j < 4; ++j) {
        v16bf b = frag_b_lds(Bs, j * 16);
        acc[0][j] = wmma_bf16(a0, b, acc[0][j]);
        acc[1][j] = wmma_bf16(a1, b, acc[1][j]);
      }
      __syncthreads();
    }
    // --- epilogue: add biases, store ---
#pragma unroll
    for (int j = 0; j < 4; ++j) {
      int gn = n0 + j * 16 + ncol;
      if (gn < Nlim) {
        float bb = isXN ? bih[H2_ + gn] : (bhh[gn] + bih[gn]);
        float* outp = isXN ? xn : hg;
        int ldo = isXN ? H_ : H3_;
#pragma unroll
        for (int s = 0; s < 2; ++s)
#pragma unroll
          for (int r = 0; r < 8; ++r) {
            int row = row0 + wave * 32 + s * 16 + half * 8 + r;
            outp[(size_t)row * ldo + gn] = acc[s][j][r] + bb;
          }
      }
    }
  }
}

// ---------------------------------------------------------------------------
// GRU gate elementwise: r,z from combined; n = tanh(xn + r*(combined_n - xn))
// ---------------------------------------------------------------------------
__global__ void gru_gate(const float* __restrict__ hg_f, const float* __restrict__ hg_r,
                         const float* __restrict__ xn_f, const float* __restrict__ xn_r,
                         float* __restrict__ h_f, float* __restrict__ h_r,
                         __bf16* __restrict__ hs_cat, int t)
{
  int idx = blockIdx.x * blockDim.x + threadIdx.x;
  if (idx >= B_ * H_) return;
  int dir = blockIdx.y;
  int b = idx / H_, j = idx % H_;
  const float* hg = dir ? hg_r : hg_f;
  const float* xn = dir ? xn_r : xn_f;
  float* h = dir ? h_r : h_f;
  int tt = dir ? (T_ - 1 - t) : t;

  float cr  = hg[(size_t)b * H3_ + j];
  float cz  = hg[(size_t)b * H3_ + H_ + j];
  float cn  = hg[(size_t)b * H3_ + 2 * H_ + j];
  float xtn = xn[(size_t)b * H_ + j];
  float r = 1.f / (1.f + __expf(-cr));
  float z = 1.f / (1.f + __expf(-cz));
  float n = tanhf(xtn + r * (cn - xtn));
  float hp = h[idx];
  float h2 = (1.f - z) * n + z * hp;
  h[idx] = h2;
  hs_cat[((size_t)tt * B_ + b) * HRS_ + dir * HSEG_ + j] = (__bf16)h2;
}

// ---------------------------------------------------------------------------
// Fused attention-score GEMM: score[row] = proj . tanh(A[row] @ W + bias)
// bf16 path: A row stride ldah with padded segments; W row for gk:
//   gk < K0 -> gk ; K0p <= gk < K0p+K1 -> K0 + (gk-K0p) ; else zero.
// f32 path: Af row stride ldaf, K0 real cols. W is K x N (x @ W convention).
// ---------------------------------------------------------------------------
__global__ __launch_bounds__(128)
void score_gemm(const __bf16* __restrict__ Ah, const float* __restrict__ Af,
                int ldah, int ldaf, int K0, int K0p, int K1,
                const float* __restrict__ W, const float* __restrict__ bias,
                const float* __restrict__ proj, int N, int KP,
                float* __restrict__ score)
{
  __shared__ __bf16 As[MT * ALD];
  __shared__ __bf16 Bs[64 * BLD];
  const int row0 = blockIdx.x * MT;
  const int tid = threadIdx.x, wave = tid >> 5;
  const int lane = tid & 31, half = lane >> 4, ncol = lane & 15;

  float rowAcc[2][8];
#pragma unroll
  for (int s = 0; s < 2; ++s)
#pragma unroll
    for (int r = 0; r < 8; ++r) rowAcc[s][r] = 0.f;

  const int ntiles = (N + 63) >> 6;
  for (int nt = 0; nt < ntiles; ++nt) {
    const int n0 = nt * 64;
    v8f acc[2][4];
#pragma unroll
    for (int s = 0; s < 2; ++s)
#pragma unroll
      for (int j = 0; j < 4; ++j) acc[s][j] = vzero8();

    for (int k0 = 0; k0 < KP; k0 += 32) {
      // --- stage A ---
#pragma unroll
      for (int it = 0; it < 4; ++it) {
        const int r = tid;
        const int gk0 = k0 + it * 8;       // scalar
        const int row = row0 + r;
        if (Ah) {
          // raw 16B copy (segments padded & aligned; stale pad lanes hit zero B rows)
          uint4 u = {0u, 0u, 0u, 0u};
          if (gk0 + 8 <= ldah)
            u = *(const uint4*)(Ah + (size_t)row * ldah + gk0);
          *(uint4*)(&As[r * ALD + it * 8]) = u;
        } else {
          float v[8];
          if (gk0 + 8 <= K0) {
#pragma unroll
            for (int i = 0; i < 8; ++i) v[i] = Af[(size_t)row * ldaf + gk0 + i];
          } else {
#pragma unroll
            for (int i = 0; i < 8; ++i) {
              int gk = gk0 + i;
              v[i] = (gk < K0) ? Af[(size_t)row * ldaf + gk] : 0.f;
            }
          }
          store8(&As[r * ALD + it * 8], v);
        }
      }
      // --- stage B (W is K x N) ---
#pragma unroll
      for (int it = 0; it < 2; ++it) {
        const int n = tid & 63;
        const int kg = __builtin_amdgcn_readfirstlane((tid >> 6) + it * 2);
        const int gk0 = k0 + kg * 8;       // scalar
        const int gn = n0 + n;
        const bool fullN = (n0 + 64 <= N);
        float v[8];
        if (fullN && gk0 + 8 <= K0) {
#pragma unroll
          for (int i = 0; i < 8; ++i) v[i] = W[(size_t)(gk0 + i) * N + gn];
        } else if (fullN && gk0 >= K0p && gk0 + 8 <= K0p + K1) {
#pragma unroll
          for (int i = 0; i < 8; ++i) v[i] = W[(size_t)(K0 + gk0 - K0p + i) * N + gn];
        } else {
#pragma unroll
          for (int i = 0; i < 8; ++i) {
            int gk = gk0 + i;
            float x = 0.f;
            if (gn < N) {
              int wrow = -1;
              if (gk < K0) wrow = gk;
              else if (gk >= K0p && gk < K0p + K1) wrow = K0 + (gk - K0p);
              if (wrow >= 0) x = W[(size_t)wrow * N + gn];
            }
            v[i] = x;
          }
        }
        store8(&Bs[n * BLD + kg * 8], v);
      }
      __syncthreads();
      v16bf a0 = frag_a_lds(As, wave * 32);
      v16bf a1 = frag_a_lds(As, wave * 32 + 16);
#pragma unroll
      for (int j = 0; j < 4; ++j) {
        v16bf b = frag_b_lds(Bs, j * 16);
        acc[0][j] = wmma_bf16(a0, b, acc[0][j]);
        acc[1][j] = wmma_bf16(a1, b, acc[1][j]);
      }
      __syncthreads();
    }
    // tanh + proj-dot, reduce over the 16 lanes of each half (same row)
#pragma unroll
    for (int j = 0; j < 4; ++j) {
      int gn = n0 + j * 16 + ncol;
      float bb = (gn < N) ? bias[gn] : 0.f;
      float pv = (gn < N) ? proj[gn] : 0.f;
#pragma unroll
      for (int s = 0; s < 2; ++s)
#pragma unroll
        for (int r = 0; r < 8; ++r) {
          float v = tanhf(acc[s][j][r] + bb) * pv;
          v += __shfl_xor(v, 1, 32);
          v += __shfl_xor(v, 2, 32);
          v += __shfl_xor(v, 4, 32);
          v += __shfl_xor(v, 8, 32);
          rowAcc[s][r] += v;
        }
    }
  }
  if (ncol == 0) {
#pragma unroll
    for (int s = 0; s < 2; ++s)
#pragma unroll
      for (int r = 0; r < 8; ++r)
        score[row0 + wave * 32 + s * 16 + half * 8 + r] = rowAcc[s][r];
  }
}

// softmax over T (axis 0) for each of B columns, in place
__global__ void softmax_time(float* __restrict__ score) {
  int b = blockIdx.x * blockDim.x + threadIdx.x;
  if (b >= B_) return;
  float mx = -1e30f;
  for (int t = 0; t < T_; ++t) mx = fmaxf(mx, score[t * B_ + b]);
  float s = 0.f;
  for (int t = 0; t < T_; ++t) { float e = __expf(score[t * B_ + b] - mx); score[t * B_ + b] = e; s += e; }
  float inv = 1.f / s;
  for (int t = 0; t < T_; ++t) score[t * B_ + b] *= inv;
}

// word_vec[b, j] = sum_t alpha[t,b] * concat(hf,hr)[t,b,j]
__global__ void word_vec_kernel(const float* __restrict__ alpha,
                                const __bf16* __restrict__ hs_cat,
                                float* __restrict__ wv)
{
  int idx = blockIdx.x * blockDim.x + threadIdx.x;
  if (idx >= B_ * H2_) return;
  int b = idx / H2_, j = idx % H2_;
  int jj = (j < H_) ? j : (HSEG_ + (j - H_));
  float s = 0.f;
  for (int t = 0; t < T_; ++t)
    s += alpha[t * B_ + b] * (float)hs_cat[((size_t)t * B_ + b) * HRS_ + jj];
  wv[idx] = s;
}

// sentence softmax (over MS consecutive rows) + weighted sum
__global__ void sent_attn(const float* __restrict__ score2,
                          const float* __restrict__ wv,
                          float* __restrict__ sv)
{
  int idx = blockIdx.x * blockDim.x + threadIdx.x;
  if (idx >= NB_ * H2_) return;
  int nb = idx / H2_, j = idx % H2_;
  float sc[MS_];
  float mx = -1e30f;
#pragma unroll
  for (int s = 0; s < MS_; ++s) { sc[s] = score2[nb * MS_ + s]; mx = fmaxf(mx, sc[s]); }
  float sum = 0.f;
#pragma unroll
  for (int s = 0; s < MS_; ++s) { sc[s] = __expf(sc[s] - mx); sum += sc[s]; }
  float inv = 1.f / sum, acc = 0.f;
#pragma unroll
  for (int s = 0; s < MS_; ++s)
    acc += sc[s] * inv * wv[((size_t)nb * MS_ + s) * H2_ + j];
  sv[idx] = acc;
}

// FC: out[512 x 53] = sv[512 x 460] @ fcW^T + fcb  (one 64-col N tile)
__global__ __launch_bounds__(128)
void fc_gemm(const float* __restrict__ A, const float* __restrict__ W,
             const float* __restrict__ bias, float* __restrict__ out)
{
  __shared__ __bf16 As[MT * ALD];
  __shared__ __bf16 Bs[64 * BLD];
  const int row0 = blockIdx.x * MT;
  const int tid = threadIdx.x, wave = tid >> 5;
  const int lane = tid & 31, half = lane >> 4, ncol = lane & 15;
  v8f acc[2][4];
#pragma unroll
  for (int s = 0; s < 2; ++s)
#pragma unroll
    for (int j = 0; j < 4; ++j) acc[s][j] = vzero8();

  for (int k0 = 0; k0 < 480; k0 += 32) {
#pragma unroll
    for (int it = 0; it < 4; ++it) {
      const int r = tid;
      const int gk0 = k0 + it * 8, row = row0 + r;
      float v[8];
      if (gk0 + 8 <= H2_) {
#pragma unroll
        for (int i = 0; i < 8; ++i) v[i] = A[(size_t)row * H2_ + gk0 + i];
      } else {
#pragma unroll
        for (int i = 0; i < 8; ++i) {
          int gk = gk0 + i;
          v[i] = (gk < H2_) ? A[(size_t)row * H2_ + gk] : 0.f;
        }
      }
      store8(&As[r * ALD + it * 8], v);
    }
#pragma unroll
    for (int it = 0; it < 2; ++it) {
      const int n = tid & 63;
      const int kg = __builtin_amdgcn_readfirstlane((tid >> 6) + it * 2);
      const int gk0 = k0 + kg * 8;
      float v[8];
#pragma unroll
      for (int i = 0; i < 8; ++i) {
        int gk = gk0 + i;
        v[i] = (n < OUT_ && gk < H2_) ? W[(size_t)n * H2_ + gk] : 0.f;
      }
      store8(&Bs[n * BLD + kg * 8], v);
    }
    __syncthreads();
    v16bf a0 = frag_a_lds(As, wave * 32);
    v16bf a1 = frag_a_lds(As, wave * 32 + 16);
#pragma unroll
    for (int j = 0; j < 4; ++j) {
      v16bf b = frag_b_lds(Bs, j * 16);
      acc[0][j] = wmma_bf16(a0, b, acc[0][j]);
      acc[1][j] = wmma_bf16(a1, b, acc[1][j]);
    }
    __syncthreads();
  }
#pragma unroll
  for (int j = 0; j < 4; ++j) {
    int gn = j * 16 + ncol;
    if (gn < OUT_) {
#pragma unroll
      for (int s = 0; s < 2; ++s)
#pragma unroll
        for (int r = 0; r < 8; ++r) {
          int row = row0 + wave * 32 + s * 16 + half * 8 + r;
          out[(size_t)row * OUT_ + gn] = acc[s][j][r] + bias[gn];
        }
    }
  }
}

__global__ void zero_f32(float* __restrict__ p, int n) {
  int i = blockIdx.x * blockDim.x + threadIdx.x;
  if (i < n) p[i] = 0.f;
}

__global__ void scatter_out(const float* __restrict__ bag_out,
                            const int* __restrict__ pairs,
                            float* __restrict__ out)
{
  int idx = blockIdx.x * blockDim.x + threadIdx.x;
  if (idx >= NB_ * OUT_) return;
  int i = idx / OUT_, o = idx % OUT_;
  int p0 = pairs[i * 3 + 0], p1 = pairs[i * 3 + 1], p2 = pairs[i * 3 + 2];
  out[(((size_t)p0 * ENT_ + p1) * ENT_ + p2) * OUT_ + o] = bag_out[(size_t)i * OUT_ + o];
}

// ---------------------------------------------------------------------------
extern "C" void kernel_launch(void* const* d_in, const int* in_sizes, int n_in,
                              void* d_out, int out_size, void* d_ws, size_t ws_size,
                              hipStream_t stream)
{
  (void)in_sizes; (void)n_in; (void)out_size; (void)ws_size;
  const float* bag   = (const float*)d_in[0];
  const float* Wih_f = (const float*)d_in[1];
  const float* Whh_f = (const float*)d_in[2];
  const float* bih_f = (const float*)d_in[3];
  const float* bhh_f = (const float*)d_in[4];
  const float* Wih_r = (const float*)d_in[5];
  const float* Whh_r = (const float*)d_in[6];
  const float* bih_r = (const float*)d_in[7];
  const float* bhh_r = (const float*)d_in[8];
  const float* Wword = (const float*)d_in[9];
  const float* bword = (const float*)d_in[10];
  const float* pword = (const float*)d_in[11];
  const float* Wsent = (const float*)d_in[12];
  const float* bsent = (const float*)d_in[13];
  const float* psent = (const float*)d_in[14];
  const float* fcW   = (const float*)d_in[15];
  const float* fcb   = (const float*)d_in[16];
  const int*   pairs = (const int*)d_in[17];
  float* out = (float*)d_out;

  char* ws = (char*)d_ws;
  size_t off = 0;
  auto alloc = [&](size_t bytes) -> void* {
    void* p = ws + off;
    off = (off + bytes + 255) & ~(size_t)255;
    return p;
  };
  float*  h_f    = (float*)alloc((size_t)B_ * H_ * 4);
  float*  h_r    = (float*)alloc((size_t)B_ * H_ * 4);
  float*  hg_f   = (float*)alloc((size_t)B_ * H3_ * 4);
  float*  hg_r   = (float*)alloc((size_t)B_ * H3_ * 4);
  float*  xn_f   = (float*)alloc((size_t)B_ * H_ * 4);
  float*  xn_r   = (float*)alloc((size_t)B_ * H_ * 4);
  __bf16* hs_cat = (__bf16*)alloc((size_t)T_ * B_ * HRS_ * 2);
  float*  score  = (float*)alloc((size_t)T_ * B_ * 4);
  float*  wv     = (float*)alloc((size_t)B_ * H2_ * 4);
  float*  score2 = (float*)alloc((size_t)B_ * 4);
  float*  sv     = (float*)alloc((size_t)NB_ * H2_ * 4);
  float*  bago   = (float*)alloc((size_t)NB_ * OUT_ * 4);

  // init hidden states and output
  zero_f32<<<(B_ * H_ + 255) / 256, 256, 0, stream>>>(h_f, B_ * H_);
  zero_f32<<<(B_ * H_ + 255) / 256, 256, 0, stream>>>(h_r, B_ * H_);
  zero_f32<<<(DOCS_ * ENT_ * ENT_ * OUT_ + 255) / 256, 256, 0, stream>>>(out, DOCS_ * ENT_ * ENT_ * OUT_);

  // sequential BiGRU: 64 steps, both directions per launch
  dim3 gemmGrid(B_ / MT, 2, 1);
  dim3 gateGrid((B_ * H_ + 255) / 256, 2, 1);
  for (int t = 0; t < T_; ++t) {
    gru_step_gemm<<<gemmGrid, 128, 0, stream>>>(bag,
        Whh_f, Wih_f, bhh_f, bih_f, Whh_r, Wih_r, bhh_r, bih_r,
        h_f, h_r, hg_f, hg_r, xn_f, xn_r, t);
    gru_gate<<<gateGrid, 256, 0, stream>>>(hg_f, hg_r, xn_f, xn_r,
                                           h_f, h_r, hs_cat, t);
  }

  // word attention scores: rows = T*B, A = concat bf16 (padded segs), K=460 (pad 480)
  score_gemm<<<(T_ * B_) / MT, 128, 0, stream>>>(hs_cat, nullptr,
      HRS_, 0, H_, HSEG_, H_, Wword, bword, pword, H2_, 480, score);
  softmax_time<<<(B_ + 255) / 256, 256, 0, stream>>>(score);
  word_vec_kernel<<<(B_ * H2_ + 255) / 256, 256, 0, stream>>>(score, hs_cat, wv);

  // sentence attention scores: rows = B, A = wv (f32)
  score_gemm<<<B_ / MT, 128, 0, stream>>>(nullptr, wv,
      0, H2_, H2_, 464, 0, Wsent, bsent, psent, H2_, 480, score2);
  sent_attn<<<(NB_ * H2_ + 255) / 256, 256, 0, stream>>>(score2, wv, sv);

  // FC + scatter
  fc_gemm<<<NB_ / MT, 128, 0, stream>>>(sv, fcW, fcb, bago);
  scatter_out<<<(NB_ * OUT_ + 255) / 256, 256, 0, stream>>>(bago, pairs, out);
}
